// DotProductAttention_88021059764554
// MI455X (gfx1250) — matrix operand order repaired and checked by reading, not compiled, not run on hardware
//
#include <hip/hip_runtime.h>

// Problem constants (match reference)
#define BB 2
#define TT 2048
#define DD 2048
#define NN 16
#define HH 128
#define BT (BB * TT)   // 4096
#define NH (NN * HH)   // 2048
#define CAPF 50.0f

typedef __attribute__((ext_vector_type(16))) __bf16 v16bf;
typedef __attribute__((ext_vector_type(8)))  __bf16 v8bf;
typedef __attribute__((ext_vector_type(8)))  float  v8f;

static __device__ __forceinline__ v8f wmma_bf16(v16bf a, v16bf b, v8f c) {
  // (neg_a, A, neg_b, B, c_mod, C, reuse_a, reuse_b)
  return __builtin_amdgcn_wmma_f32_16x16x32_bf16(false, a, false, b, (short)0, c,
                                                 false, false);
}

static __device__ __forceinline__ v8f zero8() {
  v8f z = {0.f, 0.f, 0.f, 0.f, 0.f, 0.f, 0.f, 0.f};
  return z;
}

// Branch-free CAP * tanh(x / CAP): tanh(y) = 1 - 2/(exp(2y)+1).
// __expf saturates to +inf / 0 at the extremes, giving exact +-CAP limits.
static __device__ __forceinline__ float softcap(float x) {
  const float e = __expf(x * (2.0f / CAPF));
  return CAPF * (1.0f - 2.0f / (e + 1.0f));
}

// CDNA5 async memory->LDS copy (ASYNCcnt-tracked). The LDS operand is the
// low 32 bits of the flat pointer (== group-segment byte offset).
static __device__ __forceinline__ void async_copy_b128(void* lds_dst,
                                                       const void* gsrc) {
  unsigned lds_addr = (unsigned)(uintptr_t)lds_dst;
  asm volatile("global_load_async_to_lds_b128 %0, %1, off"
               :
               : "v"(lds_addr), "v"(gsrc)
               : "memory");
}
static __device__ __forceinline__ void wait_async0() {
  asm volatile("s_wait_asynccnt 0x0" ::: "memory");
}

// ---------------------------------------------------------------------------
// QKV projection: C[bt, nh] = (A[bt,:] . W[:,nh] + bias[nh]) * scale
// A: fp32 [BT x DD] row-major. W: fp32 [DD x NH] row-major.
// VMODE 0: output bf16 [B, N, T, H]   (Q, K)
// VMODE 1: output bf16 [B, N, H, T]   (V, transposed for async PV staging)
// Block tile: 128 (M) x 64 (N), 8 waves (one 16-row strip each), K step 32.
// ---------------------------------------------------------------------------
template <int VMODE>
__global__ void __launch_bounds__(256) gemm_qkv_proj(
    const float* __restrict__ A, const float* __restrict__ W,
    const float* __restrict__ bias, __bf16* __restrict__ out, float scale) {
  __shared__ __bf16 ldsB[64 * 32];  // [n][k] transposed weight tile
  const int rowBase = blockIdx.x * 128;
  const int colBase = blockIdx.y * 64;
  const int tid  = threadIdx.x;
  const int w    = tid >> 5;
  const int lane = tid & 31;
  const int lh   = lane & 15;
  const int hfl  = lane >> 4;  // lane half (0/1)
  const int rWave = rowBase + w * 16;
  const int arow  = rWave + lh;

  v8f acc[4];
#pragma unroll
  for (int i = 0; i < 4; ++i) acc[i] = zero8();

  for (int kb = 0; kb < DD; kb += 32) {
    // Stage W tile transposed+converted: ldsB[n][k] = bf16(W[kb+k][colBase+n])
    {
      const int k    = tid >> 3;        // 0..31
      const int nOff = (tid & 7) * 8;   // 0..56
      v8f v = *(const v8f*)&W[(size_t)(kb + k) * NH + colBase + nOff];
#pragma unroll
      for (int j = 0; j < 8; ++j) ldsB[(nOff + j) * 32 + k] = (__bf16)v[j];
    }
    __syncthreads();

    // A fragment (16x32 bf16): elems 0..7 -> K=base..base+7,
    // 8..15 -> K=base+16..base+23; base = kb + 8*lane_half.
    const int abase = kb + hfl * 8;
    v8f alo = *(const v8f*)&A[(size_t)arow * DD + abase];
    v8f ahi = *(const v8f*)&A[(size_t)arow * DD + abase + 16];
    v16bf af;
#pragma unroll
    for (int e = 0; e < 8; ++e) {
      af[e]     = (__bf16)alo[e];
      af[e + 8] = (__bf16)ahi[e];
    }

    // Preload all B fragments, then issue the WMMA batch.
    v16bf bf[4];
#pragma unroll
    for (int i = 0; i < 4; ++i)
      bf[i] = __builtin_shufflevector(*(const v8bf*)&ldsB[(i * 16 + lh) * 32 + hfl * 8], *(const v8bf*)&ldsB[(i * 16 + lh) * 32 + 16 + hfl * 8], 0, 1, 2, 3, 4, 5, 6, 7, 8, 9, 10, 11, 12, 13, 14, 15);
#pragma unroll
    for (int i = 0; i < 4; ++i) acc[i] = wmma_bf16(af, bf[i], acc[i]);
    __syncthreads();
  }

  // Epilogue: C layout VGPR r: lanes0-15 M=r, lanes16-31 M=8+r; N=lane&15.
#pragma unroll
  for (int i = 0; i < 4; ++i) {
    const int col = colBase + i * 16 + lh;
    const float bv = bias[col];
    const int n = col >> 7, h = col & 127;
    if (VMODE == 0) {
#pragma unroll
      for (int r = 0; r < 8; ++r) {
        const int row = rWave + hfl * 8 + r;
        const int b = row >> 11, t = row & 2047;
        const float v = (acc[i][r] + bv) * scale;
        out[(((size_t)(b * NN + n)) * TT + t) * HH + h] = (__bf16)v;
      }
    } else {
      // rows r are consecutive t within this lane half -> 16B store.
      const int row0 = rWave + hfl * 8;            // multiple of 8
      const int b = row0 >> 11, t0 = row0 & 2047;  // 128-row tile never splits b
      v8bf vv;
#pragma unroll
      for (int r = 0; r < 8; ++r) vv[r] = (__bf16)((acc[i][r] + bv) * scale);
      *(v8bf*)&out[(((size_t)(b * NN + n)) * HH + h) * TT + t0] = vv;
    }
  }
}

// ---------------------------------------------------------------------------
// Logits: L[bn,t,s] = CAP*tanh((Q.K)/CAP); write capped logits to probs and
// accumulate per-row sum of exp (no max shift needed: |L| <= CAP so
// exp(L) in [1.9e-22, 5.2e21] and row sums stay far inside fp32 range).
// Block: one (bn, 128 t-rows); 8 waves x 16 rows; s tiles of 64.
// K tiles staged with CDNA5 async global->LDS copies.
// ---------------------------------------------------------------------------
__global__ void __launch_bounds__(256) attn_logits(
    const __bf16* __restrict__ Qp, const __bf16* __restrict__ Kp,
    float* __restrict__ probs, float* __restrict__ wsS) {
  __shared__ __bf16 ldsK[64 * 128];  // [s'][h] (straight copy)
  const int tBase = blockIdx.x * 128;
  const int bn    = blockIdx.y;
  const int tid   = threadIdx.x;
  const int w     = tid >> 5;
  const int lane  = tid & 31;
  const int lh    = lane & 15;
  const int hfl   = lane >> 4;
  const int t0    = tBase + w * 16;

  const __bf16* Qb = Qp + (size_t)bn * TT * HH;
  const __bf16* Kb = Kp + (size_t)bn * TT * HH;

  // Preload all 4 Q fragments (K=H=128); rows fixed for whole kernel.
  const int arow = t0 + lh;
  v16bf af[4];
#pragma unroll
  for (int kk = 0; kk < 4; ++kk) {
    const int abase = kk * 32 + hfl * 8;
    v8bf alo = *(const v8bf*)&Qb[(size_t)arow * HH + abase];
    v8bf ahi = *(const v8bf*)&Qb[(size_t)arow * HH + abase + 16];
#pragma unroll
    for (int e = 0; e < 8; ++e) {
      af[kk][e]     = alo[e];
      af[kk][e + 8] = ahi[e];
    }
  }

  float srow[8];
#pragma unroll
  for (int r = 0; r < 8; ++r) srow[r] = 0.f;

  for (int sT = 0; sT < TT; sT += 64) {
    // Async-stage 64 K rows (64x128 bf16): 4 x b128 per thread.
#pragma unroll
    for (int c = 0; c < 4; ++c) {
      const int e0 = (c * 256 + tid) * 8;
      const int sp = e0 >> 7, h0 = e0 & 127;
      async_copy_b128(&ldsK[sp * 128 + h0],
                      &Kb[(size_t)(sT + sp) * HH + h0]);
    }
    wait_async0();
    __syncthreads();

#pragma unroll
    for (int i = 0; i < 4; ++i) {
      v16bf bf[4];
#pragma unroll
      for (int kk = 0; kk < 4; ++kk)
        bf[kk] = __builtin_shufflevector(*(const v8bf*)&ldsK[(i * 16 + lh) * 128 + kk * 32 + hfl * 8], *(const v8bf*)&ldsK[(i * 16 + lh) * 128 + kk * 32 + 16 + hfl * 8], 0, 1, 2, 3, 4, 5, 6, 7, 8, 9, 10, 11, 12, 13, 14, 15);
      v8f lg = zero8();
#pragma unroll
      for (int kk = 0; kk < 4; ++kk) lg = wmma_bf16(af[kk], bf[kk], lg);

      const int scol = sT + i * 16 + lh;
#pragma unroll
      for (int r = 0; r < 8; ++r) {
        const float x = softcap(lg[r]);
        const int t = t0 + hfl * 8 + r;
        probs[((size_t)(bn * TT + t)) * TT + scol] = x;
        srow[r] += __expf(x);
      }
    }
    __syncthreads();
  }

  // Sum-reduce across the 16 lanes holding each row's columns.
#pragma unroll
  for (int r = 0; r < 8; ++r) {
    float s = srow[r];
#pragma unroll
    for (int off = 1; off < 16; off <<= 1) s += __shfl_xor(s, off, 32);
    if (lh == 0) {
      const int t = t0 + hfl * 8 + r;
      wsS[bn * TT + t] = s;
    }
  }
}

// ---------------------------------------------------------------------------
// PV: p = exp(L)/sum (overwrite probs with final softmax), enc = P.V (WMMA).
// V is stored transposed [B,N,H,T] -> LDS tile is a straight async copy.
// Each lane owns one P row -> scalar sum.
// ---------------------------------------------------------------------------
__global__ void __launch_bounds__(256) attn_pv(
    const __bf16* __restrict__ Vt, float* __restrict__ probs,
    const float* __restrict__ wsS, __bf16* __restrict__ enc) {
  __shared__ __bf16 ldsV[128 * 32];  // [h][k]
  const int tBase = blockIdx.x * 128;
  const int bn    = blockIdx.y;
  const int tid   = threadIdx.x;
  const int w     = tid >> 5;
  const int lane  = tid & 31;
  const int lh    = lane & 15;
  const int hfl   = lane >> 4;
  const int t0    = tBase + w * 16;

  const __bf16* Vb = Vt + (size_t)bn * HH * TT;  // [h][t]
  const int arow = t0 + lh;  // this lane's P row
  const float rinv = 1.0f / wsS[bn * TT + arow];
  float* prow = probs + ((size_t)(bn * TT + arow)) * TT;

  v8f acc[8];
#pragma unroll
  for (int i = 0; i < 8; ++i) acc[i] = zero8();

  for (int s0 = 0; s0 < TT; s0 += 32) {
    // Async-stage V tile [128 h][32 s]: 2 x b128 per thread, straight copy.
#pragma unroll
    for (int c = 0; c < 2; ++c) {
      const int e0 = (c * 256 + tid) * 8;
      const int h = e0 >> 5, k0 = e0 & 31;
      async_copy_b128(&ldsV[h * 32 + k0], &Vb[(size_t)h * TT + s0 + k0]);
    }
    wait_async0();
    __syncthreads();

    // P fragment directly from global in A-fragment pattern; normalize,
    // write final softmax back, convert to bf16.
    const int base = s0 + hfl * 8;
    v8f plo = *(const v8f*)&prow[base];
    v8f phi = *(const v8f*)&prow[base + 16];
    v16bf pa;
#pragma unroll
    for (int e = 0; e < 8; ++e) {
      const float p0 = __expf(plo[e]) * rinv;
      const float p1 = __expf(phi[e]) * rinv;
      plo[e] = p0;
      phi[e] = p1;
      pa[e]     = (__bf16)p0;
      pa[e + 8] = (__bf16)p1;
    }
    *(v8f*)&prow[base]      = plo;
    *(v8f*)&prow[base + 16] = phi;

    v16bf bf[8];
#pragma unroll
    for (int i = 0; i < 8; ++i)
      bf[i] = __builtin_shufflevector(*(const v8bf*)&ldsV[(i * 16 + lh) * 32 + hfl * 8], *(const v8bf*)&ldsV[(i * 16 + lh) * 32 + 16 + hfl * 8], 0, 1, 2, 3, 4, 5, 6, 7, 8, 9, 10, 11, 12, 13, 14, 15);
#pragma unroll
    for (int i = 0; i < 8; ++i) acc[i] = wmma_bf16(pa, bf[i], acc[i]);
    __syncthreads();
  }

  // enc [B,T,N,H] bf16
  const int b = bn >> 4, n = bn & 15;
#pragma unroll
  for (int i = 0; i < 8; ++i) {
    const int h = i * 16 + lh;
#pragma unroll
    for (int r = 0; r < 8; ++r) {
      const int t = t0 + hfl * 8 + r;
      enc[(((size_t)(b * TT + t)) * NN + n) * HH + h] = (__bf16)acc[i][r];
    }
  }
}

// ---------------------------------------------------------------------------
// Output projection: out[bt,d] = enc[bt,:] . wo[d,:] + bo[d]
// enc: bf16 [BT x NH] row-major. wo: fp32 [DD x NH] (transpose is contiguous
// along K for fixed d). out fp32.
// ---------------------------------------------------------------------------
__global__ void __launch_bounds__(256) gemm_out(
    const __bf16* __restrict__ A, const float* __restrict__ Wo,
    const float* __restrict__ bo, float* __restrict__ out) {
  __shared__ __bf16 ldsB[64 * 32];  // [d'][k]
  const int rowBase = blockIdx.x * 128;
  const int colBase = blockIdx.y * 64;
  const int tid  = threadIdx.x;
  const int w    = tid >> 5;
  const int lane = tid & 31;
  const int lh   = lane & 15;
  const int hfl  = lane >> 4;
  const int rWave = rowBase + w * 16;
  const int arow  = rWave + lh;

  v8f acc[4];
#pragma unroll
  for (int i = 0; i < 4; ++i) acc[i] = zero8();

  for (int kb = 0; kb < NH; kb += 32) {
    // Stage Wo^T tile converted: ldsB[d'][k] = bf16(Wo[colBase+d'][kb+k])
    {
      const int np   = tid >> 2;       // 0..63
      const int kOff = (tid & 3) * 8;  // 0,8,16,24
      v8f v = *(const v8f*)&Wo[(size_t)(colBase + np) * NH + kb + kOff];
      v8bf vb;
#pragma unroll
      for (int j = 0; j < 8; ++j) vb[j] = (__bf16)v[j];
      *(v8bf*)&ldsB[np * 32 + kOff] = vb;
    }
    __syncthreads();

    const int abase = kb + hfl * 8;
    v8bf alo = *(const v8bf*)&A[(size_t)arow * NH + abase];
    v8bf ahi = *(const v8bf*)&A[(size_t)arow * NH + abase + 16];
    v16bf af;
#pragma unroll
    for (int e = 0; e < 8; ++e) {
      af[e]     = alo[e];
      af[e + 8] = ahi[e];
    }

    v16bf bf[4];
#pragma unroll
    for (int i = 0; i < 4; ++i)
      bf[i] = __builtin_shufflevector(*(const v8bf*)&ldsB[(i * 16 + lh) * 32 + hfl * 8], *(const v8bf*)&ldsB[(i * 16 + lh) * 32 + 16 + hfl * 8], 0, 1, 2, 3, 4, 5, 6, 7, 8, 9, 10, 11, 12, 13, 14, 15);
#pragma unroll
    for (int i = 0; i < 4; ++i) acc[i] = wmma_bf16(af, bf[i], acc[i]);
    __syncthreads();
  }

#pragma unroll
  for (int i = 0; i < 4; ++i) {
    const int col = colBase + i * 16 + lh;
    const float bv = bo[col];
#pragma unroll
    for (int r = 0; r < 8; ++r) {
      const int row = rWave + hfl * 8 + r;
      out[(size_t)row * DD + col] = acc[i][r] + bv;
    }
  }
}

// ---------------------------------------------------------------------------
extern "C" void kernel_launch(void* const* d_in, const int* in_sizes, int n_in,
                              void* d_out, int out_size, void* d_ws,
                              size_t ws_size, hipStream_t stream) {
  const float* qv = (const float*)d_in[0];
  const float* kv = (const float*)d_in[1];
  const float* vv = (const float*)d_in[2];
  const float* wq = (const float*)d_in[3];
  const float* bq = (const float*)d_in[4];
  const float* wk = (const float*)d_in[5];
  const float* bk = (const float*)d_in[6];
  const float* wv = (const float*)d_in[7];
  const float* bv = (const float*)d_in[8];
  const float* wo = (const float*)d_in[9];
  const float* bo = (const float*)d_in[10];

  // Workspace layout (bf16 region then fp32 row-sums):
  const size_t QPN = (size_t)BB * NN * TT * HH;  // 8,388,608 elems
  __bf16* Qp  = (__bf16*)d_ws;
  __bf16* Kp  = Qp + QPN;
  __bf16* Vp  = Kp + QPN;   // stored transposed [B,N,H,T]
  __bf16* enc = Vp + QPN;
  float* wsS = (float*)(enc + QPN);

  float* out   = (float*)d_out;
  float* probs = out + (size_t)BT * DD;

  const float qscale = 0.08838834764831845f;  // H^-0.5, H=128

  dim3 gP(BT / 128, NH / 64);  // 32 x 32
  gemm_qkv_proj<0><<<gP, 256, 0, stream>>>(qv, wq, bq, Qp, qscale);
  gemm_qkv_proj<0><<<gP, 256, 0, stream>>>(kv, wk, bk, Kp, 1.0f);
  gemm_qkv_proj<1><<<gP, 256, 0, stream>>>(vv, wv, bv, Vp, 1.0f);

  dim3 gA(TT / 128, BB * NN);  // 16 x 32
  attn_logits<<<gA, 256, 0, stream>>>(Qp, Kp, probs, wsS);
  attn_pv<<<gA, 256, 0, stream>>>(Vp, probs, wsS, enc);

  dim3 gO(BT / 128, DD / 64);  // 32 x 32
  gemm_out<<<gO, 256, 0, stream>>>(enc, wo, bo, out);
}
